// RouteNetModel_3272765079859
// MI455X (gfx1250) — compile-verified, matrix operand order checked by reading, not run on hardware
//
#include <hip/hip_runtime.h>

typedef __attribute__((ext_vector_type(16))) _Float16 v16h;
typedef __attribute__((ext_vector_type(8)))  _Float16 v8h;
typedef __attribute__((ext_vector_type(8)))  float    v8f;

// K index held by (lane, element) in the 16-bit WMMA A/B operand layout
// (CDNA5 ISA 7.12.2). Per lane the 16 elements cover two contiguous K runs:
// [off, off+8) and [off+16, off+24) with off = (lane&16) ? 8 : 0.
__device__ __forceinline__ int kmap(int lane, int e) {
    return (e < 8 ? e : e + 8) + ((lane & 16) ? 8 : 0);
}

__device__ __forceinline__ v8f wmma16(v16h a, v16h b, v8f c) {
    return __builtin_amdgcn_wmma_f32_16x16x32_f16(false, a, false, b, (short)0, c,
                                                  false, false);
}

__device__ __forceinline__ float fast_rcp(float x) { return __builtin_amdgcn_rcpf(x); }
__device__ __forceinline__ float sigm(float x) {
    return fast_rcp(1.0f + __expf(-x));          // v_exp + v_rcp, no div chain
}
__device__ __forceinline__ float tanh_fast(float x) {
    return 1.0f - 2.0f * fast_rcp(1.0f + __expf(2.0f * x));
}
__device__ __forceinline__ float selu(float x) {
    const float a = 1.6732632423543772f, s = 1.0507009873554805f;
    return x > 0.f ? s * x : s * a * (__expf(x) - 1.f);
}

// pack 16 f32 (two contiguous K runs) into a WMMA 16-half operand
__device__ __forceinline__ v16h cvt16(float4 a, float4 b, float4 c, float4 d) {
    v16h r;
    r[0]  = (_Float16)a.x; r[1]  = (_Float16)a.y; r[2]  = (_Float16)a.z; r[3]  = (_Float16)a.w;
    r[4]  = (_Float16)b.x; r[5]  = (_Float16)b.y; r[6]  = (_Float16)b.z; r[7]  = (_Float16)b.w;
    r[8]  = (_Float16)c.x; r[9]  = (_Float16)c.y; r[10] = (_Float16)c.z; r[11] = (_Float16)c.w;
    r[12] = (_Float16)d.x; r[13] = (_Float16)d.y; r[14] = (_Float16)d.z; r[15] = (_Float16)d.w;
    return r;
}

__device__ __forceinline__ v16h cat8(v8h lo, v8h hi) {
    return __builtin_shufflevector(lo, hi, 0, 1, 2, 3, 4, 5, 6, 7,
                                   8, 9, 10, 11, 12, 13, 14, 15);
}

// load a 16-row x 32-K f32 tile row (per-lane) as an A operand
__device__ __forceinline__ v16h loadA_f32(const float* __restrict__ row, int off) {
    const float4* p = (const float4*)(row + off);
    const float4* q = (const float4*)(row + off + 16);
    return cvt16(p[0], p[1], q[0], q[1]);
}

// ---------------------------------------------------------------- utilities
__global__ void zero_f32(float* p, int n) {
    int i = blockIdx.x * blockDim.x + threadIdx.x;
    if (i < n) p[i] = 0.f;
}

__global__ void init_states(const float* cap, const float* traf,
                            float* link_state, float* path_state,
                            int n_links, int n_paths) {
    int i = blockIdx.x * blockDim.x + threadIdx.x;
    int nl = n_links * 32;
    int np = n_paths * 32;
    if (i < nl) {
        link_state[i] = ((i & 31) == 0) ? cap[i >> 5] : 0.f;
    } else if (i < nl + np) {
        int j = i - nl;
        path_state[j] = ((j & 31) == 0) ? traf[j >> 5] : 0.f;
    }
}

__global__ void calc_meta(const int* paths, const int* seqs,
                          int* starts, int* lens, int n_total) {
    int i = blockIdx.x * blockDim.x + threadIdx.x;
    if (i >= n_total) return;
    atomicAdd(&lens[paths[i]], 1);
    if (seqs[i] == 0) starts[paths[i]] = i;
}

// Pack a row-major [K x N] f32 matrix into WMMA B-operand f16 tiles.
__global__ void pack_b(const float* src, _Float16* dst, int K, int N) {
    int idx = blockIdx.x * blockDim.x + threadIdx.x;
    int ntn = N / 16, ntk = K / 32;
    int total = ntk * ntn * 32 * 16;
    if (idx >= total) return;
    int e    = idx & 15;
    int lane = (idx >> 4) & 31;
    int tile = idx >> 9;
    int tn   = tile % ntn;
    int tk   = tile / ntn;
    int k = tk * 32 + kmap(lane, e);
    int n = tn * 16 + (lane & 15);
    dst[idx] = (_Float16)src[k * N + n];
}

// ---------------------------------------------------------------- path RNN
// One wave32 per 16-path tile. GRU state kept in C-layout registers across the
// whole sequence; per-step output scattered into m_link via atomicAdd
// (== unsorted segment_sum over links).
__global__ __launch_bounds__(32)
void path_rnn(const float* __restrict__ link_state, float* __restrict__ path_state,
              float* __restrict__ m_link, const int* __restrict__ links,
              const int* __restrict__ starts, const int* __restrict__ lens,
              const _Float16* __restrict__ pW, const _Float16* __restrict__ pU,
              const float* __restrict__ pb, int n_paths) {
    const int p0   = blockIdx.x * 16;
    const int lane = threadIdx.x;
    const int ncol = lane & 15;
    const int mb   = (lane >> 4) << 3;     // C-layout row base: 0 or 8
    const int mA   = lane & 15;            // A-layout row
    const int off  = (lane & 16) ? 8 : 0;  // K run base

    const int pA      = p0 + mA;
    const bool pValid = pA < n_paths;
    const int startA  = pValid ? starts[pA] : 0;
    const int lenA    = pValid ? lens[pA] : 0;

    // tile max length (uniform across wave)
    int tl = lenA;
    #pragma unroll
    for (int d = 1; d < 16; d <<= 1) tl = max(tl, __shfl_xor(tl, d));

    float bias[6];
    #pragma unroll
    for (int j = 0; j < 6; ++j) bias[j] = pb[j * 16 + ncol];

    // per-row metadata for the 8 C-layout rows this lane owns
    int startR[8], lenR[8];
    #pragma unroll
    for (int i = 0; i < 8; ++i) {
        int p = p0 + mb + i;
        bool v = p < n_paths;
        startR[i] = v ? starts[p] : 0;
        lenR[i]   = v ? lens[p] : 0;
    }

    // load h in C-layout (dims 0-15 -> h0, 16-31 -> h1)
    v8f h0, h1;
    #pragma unroll
    for (int i = 0; i < 8; ++i) {
        int p = p0 + mb + i;
        bool v = p < n_paths;
        h0[i] = v ? path_state[p * 32 + ncol] : 0.f;
        h1[i] = v ? path_state[p * 32 + 16 + ncol] : 0.f;
    }

    __shared__ _Float16 hst[16 * 32];

    for (int t = 0; t < tl; ++t) {
        // per-row link id: unconditional clamped load (no exec branching),
        // then select -1 for rows past their length.
        int L[8];
        #pragma unroll
        for (int i = 0; i < 8; ++i) {
            int tc = min(t, lenR[i] - 1);       // may be -1 when lenR==0
            int a  = startR[i] + max(tc, 0);    // always in-bounds
            int v  = links[a];
            L[i] = (t < lenR[i]) ? v : -1;
        }

        __syncthreads();
        #pragma unroll
        for (int i = 0; i < 8; ++i) {
            hst[(mb + i) * 32 + ncol]      = (_Float16)h0[i];
            hst[(mb + i) * 32 + 16 + ncol] = (_Float16)h1[i];
        }
        __syncthreads();

        // h as A operand: two contiguous 16B LDS loads
        v8h hlo = *(const v8h*)&hst[mA * 32 + off];
        v8h hhi = *(const v8h*)&hst[mA * 32 + off + 16];
        v16h ha = cat8(hlo, hhi);

        // x = link_state[link at (path mA, step t)]; rows past len produce
        // garbage that WMMA computes but the masked update discards, so the
        // load is unconditional with a clamped row index.
        int lrow = (t < lenA) ? links[startA + min(t, max(lenA - 1, 0))] : 0;
        v16h xa = loadA_f32(link_state + (size_t)lrow * 32, off);

        v8f gx[6], gh[6];
        #pragma unroll
        for (int j = 0; j < 6; ++j) {
            v16h Bw = *(const v16h*)(pW + (j * 32 + lane) * 16);
            v16h Bu = *(const v16h*)(pU + (j * 32 + lane) * 16);
            v8f z = {};
            gx[j] = wmma16(xa, Bw, z);
            gh[j] = wmma16(ha, Bu, z);
        }

        #pragma unroll
        for (int i = 0; i < 8; ++i) {
            if (L[i] >= 0) {
                float z0 = sigm(gx[0][i] + bias[0] + gh[0][i]);
                float r0 = sigm(gx[2][i] + bias[2] + gh[2][i]);
                float c0 = tanh_fast(gx[4][i] + bias[4] + r0 * gh[4][i]);
                float n0 = z0 * h0[i] + (1.f - z0) * c0;

                float z1 = sigm(gx[1][i] + bias[1] + gh[1][i]);
                float r1 = sigm(gx[3][i] + bias[3] + gh[3][i]);
                float c1 = tanh_fast(gx[5][i] + bias[5] + r1 * gh[5][i]);
                float n1 = z1 * h1[i] + (1.f - z1) * c1;

                h0[i] = n0;
                h1[i] = n1;
                atomicAdd(&m_link[L[i] * 32 + ncol], n0);
                atomicAdd(&m_link[L[i] * 32 + 16 + ncol], n1);
            }
        }
    }

    #pragma unroll
    for (int i = 0; i < 8; ++i) {
        int p = p0 + mb + i;
        if (p < n_paths) {
            path_state[p * 32 + ncol]      = h0[i];
            path_state[p * 32 + 16 + ncol] = h1[i];
        }
    }
}

// ---------------------------------------------------------------- edge GRU
__global__ __launch_bounds__(32)
void edge_gru(float* __restrict__ link_state, const float* __restrict__ m_link,
              const _Float16* __restrict__ eW, const _Float16* __restrict__ eU,
              const float* __restrict__ eb, int n_links) {
    const int l0   = blockIdx.x * 16;
    const int lane = threadIdx.x;
    const int ncol = lane & 15;
    const int mb   = (lane >> 4) << 3;
    const int mA   = lane & 15;
    const int off  = (lane & 16) ? 8 : 0;

    __shared__ _Float16 hst[16 * 32];
    v8f h0, h1;
    #pragma unroll
    for (int i = 0; i < 8; ++i) {
        int r = l0 + mb + i;
        bool v = r < n_links;
        h0[i] = v ? link_state[r * 32 + ncol] : 0.f;
        h1[i] = v ? link_state[r * 32 + 16 + ncol] : 0.f;
        hst[(mb + i) * 32 + ncol]      = (_Float16)h0[i];
        hst[(mb + i) * 32 + 16 + ncol] = (_Float16)h1[i];
    }
    __syncthreads();

    v8h hlo = *(const v8h*)&hst[mA * 32 + off];
    v8h hhi = *(const v8h*)&hst[mA * 32 + off + 16];
    v16h ha = cat8(hlo, hhi);

    int xrow = min(l0 + mA, n_links - 1);
    v16h xa = loadA_f32(m_link + (size_t)xrow * 32, off);

    float bias[6];
    #pragma unroll
    for (int j = 0; j < 6; ++j) bias[j] = eb[j * 16 + ncol];

    v8f gx[6], gh[6];
    #pragma unroll
    for (int j = 0; j < 6; ++j) {
        v16h Bw = *(const v16h*)(eW + (j * 32 + lane) * 16);
        v16h Bu = *(const v16h*)(eU + (j * 32 + lane) * 16);
        v8f z = {};
        gx[j] = wmma16(xa, Bw, z);
        gh[j] = wmma16(ha, Bu, z);
    }

    #pragma unroll
    for (int i = 0; i < 8; ++i) {
        int r = l0 + mb + i;
        if (r >= n_links) continue;
        float z0 = sigm(gx[0][i] + bias[0] + gh[0][i]);
        float r0 = sigm(gx[2][i] + bias[2] + gh[2][i]);
        float c0 = tanh_fast(gx[4][i] + bias[4] + r0 * gh[4][i]);
        float z1 = sigm(gx[1][i] + bias[1] + gh[1][i]);
        float r1 = sigm(gx[3][i] + bias[3] + gh[3][i]);
        float c1 = tanh_fast(gx[5][i] + bias[5] + r1 * gh[5][i]);
        link_state[r * 32 + ncol]      = z0 * h0[i] + (1.f - z0) * c0;
        link_state[r * 32 + 16 + ncol] = z1 * h1[i] + (1.f - z1) * c1;
    }
}

// ---------------------------------------------------------------- readout
// Fused 32->256 selu ->256 selu ->1 per 16-path tile, one wave per block.
__global__ __launch_bounds__(32)
void readout(const float* __restrict__ path_state,
             const _Float16* __restrict__ rW1, const float* __restrict__ rb1,
             const _Float16* __restrict__ rW2, const float* __restrict__ rb2,
             const float* __restrict__ rW3, const float* __restrict__ rb3,
             float* __restrict__ out, int n_paths) {
    const int p0   = blockIdx.x * 16;
    const int lane = threadIdx.x;
    const int ncol = lane & 15;
    const int mb   = (lane >> 4) << 3;
    const int mA   = lane & 15;
    const int off  = (lane & 16) ? 8 : 0;

    __shared__ _Float16 H1[16 * 256];   // 8 KB

    int prow = min(p0 + mA, n_paths - 1);
    v16h xa = loadA_f32(path_state + (size_t)prow * 32, off);

    // layer 1: 16 N-tiles of W1 (32x256)
    #pragma unroll
    for (int nt = 0; nt < 16; ++nt) {
        v16h B = *(const v16h*)(rW1 + (nt * 32 + lane) * 16);
        v8f z = {};
        v8f acc = wmma16(xa, B, z);
        float bcol = rb1[nt * 16 + ncol];
        #pragma unroll
        for (int i = 0; i < 8; ++i)
            H1[(mb + i) * 256 + nt * 16 + ncol] = (_Float16)selu(acc[i] + bcol);
    }
    __syncthreads();

    // layer 2: H1 (16x256) @ W2 (256x256), K in 8 chunks of 32
    v8f acc2[16];
    #pragma unroll
    for (int nt = 0; nt < 16; ++nt) { v8f z = {}; acc2[nt] = z; }
    #pragma unroll
    for (int kt = 0; kt < 8; ++kt) {
        v8h alo = *(const v8h*)&H1[mA * 256 + kt * 32 + off];
        v8h ahi = *(const v8h*)&H1[mA * 256 + kt * 32 + off + 16];
        v16h A2 = cat8(alo, ahi);
        #pragma unroll
        for (int nt = 0; nt < 16; ++nt) {
            v16h B = *(const v16h*)(rW2 + ((kt * 16 + nt) * 32 + lane) * 16);
            acc2[nt] = wmma16(A2, B, acc2[nt]);
        }
    }

    // layer 3: selu + dot with W3 (256x1)
    float part[8];
    #pragma unroll
    for (int i = 0; i < 8; ++i) part[i] = 0.f;
    #pragma unroll
    for (int nt = 0; nt < 16; ++nt) {
        float bcol = rb2[nt * 16 + ncol];
        float w3   = rW3[nt * 16 + ncol];
        #pragma unroll
        for (int i = 0; i < 8; ++i)
            part[i] += selu(acc2[nt][i] + bcol) * w3;
    }
    float b3 = rb3[0];
    #pragma unroll
    for (int i = 0; i < 8; ++i) {
        float s = part[i];
        s += __shfl_xor(s, 1);
        s += __shfl_xor(s, 2);
        s += __shfl_xor(s, 4);
        s += __shfl_xor(s, 8);
        if (ncol == 0) {
            int p = p0 + mb + i;
            if (p < n_paths) out[p] = s + b3;
        }
    }
}

// ---------------------------------------------------------------- host
extern "C" void kernel_launch(void* const* d_in, const int* in_sizes, int n_in,
                              void* d_out, int out_size, void* d_ws, size_t ws_size,
                              hipStream_t stream) {
    const float* cap    = (const float*)d_in[0];
    const float* traf   = (const float*)d_in[1];
    const float* path_W = (const float*)d_in[2];
    const float* path_U = (const float*)d_in[3];
    const float* path_b = (const float*)d_in[4];
    const float* edge_W = (const float*)d_in[5];
    const float* edge_U = (const float*)d_in[6];
    const float* edge_b = (const float*)d_in[7];
    const float* ro_W1  = (const float*)d_in[8];
    const float* ro_b1  = (const float*)d_in[9];
    const float* ro_W2  = (const float*)d_in[10];
    const float* ro_b2  = (const float*)d_in[11];
    const float* ro_W3  = (const float*)d_in[12];
    const float* ro_b3  = (const float*)d_in[13];
    const int* links  = (const int*)d_in[14];
    const int* pathsA = (const int*)d_in[15];
    const int* seqsA  = (const int*)d_in[16];
    const int n_links = in_sizes[0];
    const int n_paths = in_sizes[1];
    const int n_total = in_sizes[14];
    (void)n_in; (void)out_size; (void)ws_size;

    char* w = (char*)d_ws;
    auto carve = [&](size_t bytes) {
        char* p = w;
        w += (bytes + 255) & ~(size_t)255;
        return p;
    };
    float* link_state = (float*)carve((size_t)n_links * 32 * 4);
    float* path_state = (float*)carve((size_t)n_paths * 32 * 4);
    float* m_link     = (float*)carve((size_t)n_links * 32 * 4);
    int*   starts     = (int*)carve((size_t)n_paths * 4);
    int*   lens       = (int*)carve((size_t)n_paths * 4);
    _Float16* pWp  = (_Float16*)carve(3072 * 2);
    _Float16* pUp  = (_Float16*)carve(3072 * 2);
    _Float16* eWp  = (_Float16*)carve(3072 * 2);
    _Float16* eUp  = (_Float16*)carve(3072 * 2);
    _Float16* rW1p = (_Float16*)carve(8192 * 2);
    _Float16* rW2p = (_Float16*)carve(65536 * 2);

    int tot_init = (n_links + n_paths) * 32;
    init_states<<<(tot_init + 255) / 256, 256, 0, stream>>>(cap, traf, link_state,
                                                            path_state, n_links, n_paths);
    zero_f32<<<(n_paths + 255) / 256, 256, 0, stream>>>((float*)lens, n_paths);
    calc_meta<<<(n_total + 255) / 256, 256, 0, stream>>>(pathsA, seqsA, starts, lens,
                                                         n_total);
    pack_b<<<(3072 + 255) / 256, 256, 0, stream>>>(path_W, pWp, 32, 96);
    pack_b<<<(3072 + 255) / 256, 256, 0, stream>>>(path_U, pUp, 32, 96);
    pack_b<<<(3072 + 255) / 256, 256, 0, stream>>>(edge_W, eWp, 32, 96);
    pack_b<<<(3072 + 255) / 256, 256, 0, stream>>>(edge_U, eUp, 32, 96);
    pack_b<<<(8192 + 255) / 256, 256, 0, stream>>>(ro_W1, rW1p, 32, 256);
    pack_b<<<(65536 + 255) / 256, 256, 0, stream>>>(ro_W2, rW2p, 256, 256);

    for (int it = 0; it < 8; ++it) {
        zero_f32<<<(n_links * 32 + 255) / 256, 256, 0, stream>>>(m_link, n_links * 32);
        path_rnn<<<(n_paths + 15) / 16, 32, 0, stream>>>(
            link_state, path_state, m_link, links, starts, lens, pWp, pUp, path_b,
            n_paths);
        edge_gru<<<(n_links + 15) / 16, 32, 0, stream>>>(link_state, m_link, eWp, eUp,
                                                         edge_b, n_links);
    }

    readout<<<(n_paths + 15) / 16, 32, 0, stream>>>(path_state, rW1p, ro_b1, rW2p,
                                                    ro_b2, ro_W3, ro_b3, (float*)d_out,
                                                    n_paths);
}
